// MultiHeadAttention_11364483465630
// MI455X (gfx1250) — matrix operand order repaired and checked by reading, not compiled, not run on hardware
//
#include <hip/hip_runtime.h>
#include <hip/hip_bf16.h>

// ---------------- problem constants ----------------
constexpr int B_  = 16;
constexpr int S_  = 1024;
constexpr int D_  = 512;
constexpr int H_  = 8;
constexpr int DK_ = 64;

// ---------------- WMMA types ----------------
typedef __attribute__((ext_vector_type(16))) __bf16 v16bf;
typedef __attribute__((ext_vector_type(8)))  __bf16 v8bf;
typedef __attribute__((ext_vector_type(8)))  float  v8f;

#define DEVI __device__ __forceinline__

DEVI v8f wmma_bf16(v16bf a, v16bf b, v8f c) {
  // D = A(16x32) * B(32x16) + C(16x16), f32 accumulate
  return __builtin_amdgcn_wmma_f32_16x16x32_bf16(
      /*neg_a=*/false, a, /*neg_b=*/false, b,
      /*c_mod=*/(short)0, c, /*reuse_a=*/false, /*reuse_b=*/false);
}

// ---- A fragment (16x32 bf16): row M = lane&15
//      elems 0..15 -> K = (lane>>4)*16 + 0..15
// Two aligned 16-byte vector loads + shuffle (no VALU conversion).
DEVI v16bf load_a16(const __bf16* base, size_t ld, int lane) {
  const __bf16* p = base + (size_t)(lane & 15) * ld + ((lane >> 4) << 4);
  v8bf lo = *(const v8bf*)(p);
  v8bf hi = *(const v8bf*)(p + 8);
  return __builtin_shufflevector(lo, hi, 0, 1, 2, 3, 4, 5, 6, 7,
                                 8, 9, 10, 11, 12, 13, 14, 15);
}

// ---- B fragment (32x16): col N = lane&15, K = (lane>>4)*16 + e,
// where B(k,n) = base[n*ld + k]  (source holds the K-dim contiguously,
// i.e. K^T / transposed weights / transposed V).  One 32-byte load.
DEVI v16bf load_bT(const __bf16* base, size_t ld, int lane) {
  const __bf16* p = base + (size_t)(lane & 15) * ld + (((lane >> 4) & 1) << 4);
  return *(const v16bf*)p;
}

// =====================================================================
// Kernel 0a: elementwise f32 -> bf16 (8 elems / thread, vector ld/st)
// =====================================================================
__global__ __launch_bounds__(256) void cvt_kernel(const float* __restrict__ in,
                                                  __bf16* __restrict__ out) {
  const size_t i = ((size_t)blockIdx.x * blockDim.x + threadIdx.x) * 8;
  v8f x = *(const v8f*)(in + i);
  *(v8bf*)(out + i) = __builtin_convertvector(x, v8bf);
}

// =====================================================================
// Kernel 0b: transpose + convert  W[batch][rows][cols] -> WT[batch][cols][rows]
// =====================================================================
__global__ __launch_bounds__(256) void wtrans_kernel(const float* __restrict__ in,
                                                     __bf16* __restrict__ out,
                                                     int rows, int cols) {
  const size_t idx = (size_t)blockIdx.x * blockDim.x + threadIdx.x;
  const size_t mat = (size_t)rows * cols;
  const size_t bat = idx / mat;                 // batch (head or 0)
  const size_t r2  = idx % mat;                 // output-linear: [col][row]
  const size_t c   = r2 / rows;                 // output row   = input col
  const size_t r   = r2 % rows;                 // output col   = input row
  out[idx] = (__bf16)in[bat * mat + r * cols + c];
}

// =====================================================================
// Kernel 1: QKV projections from pre-converted bf16 activations/weights.
// One wave -> one 16-row x 64-col (DK) tile of one projection.
// WT layout: [H][DK][D] so B fragments are contiguous vector loads.
// V projection is stored transposed: Vt[B,H][DK][S].
// =====================================================================
__global__ __launch_bounds__(256) void qkv_proj_kernel(
    const __bf16* __restrict__ Xq, const __bf16* __restrict__ Xv,
    const __bf16* __restrict__ WqT, const __bf16* __restrict__ WkT,
    const __bf16* __restrict__ WvT,
    const float* __restrict__ bq, const float* __restrict__ bk,
    const float* __restrict__ bv,
    __bf16* __restrict__ Qp, __bf16* __restrict__ Kp, __bf16* __restrict__ Vt) {
  const int lane = threadIdx.x & 31;
  const int wave = threadIdx.x >> 5;
  const int gid  = blockIdx.x * 8 + wave;

  const int proj  = gid >> 13;            // 8192 wave-tiles per projection
  const int rem   = gid & 8191;
  const int b     = rem >> 9;             // / (H_ * 64)
  const int rem2  = rem & 511;
  const int h     = rem2 >> 6;
  const int stile = rem2 & 63;

  const __bf16* X    = (proj == 0) ? Xq : Xv;
  const __bf16* WT   = (proj == 0) ? WqT : (proj == 1) ? WkT : WvT;
  const float*  bias = (proj == 0) ? bq : (proj == 1) ? bk : bv;

  const __bf16* Xbase = X + ((size_t)b * S_ + (size_t)stile * 16) * D_;
  const __bf16* WTh   = WT + (size_t)h * DK_ * D_;   // [DK][D]

  v8f acc[4] = {};                         // 4 dk-column tiles of 16
  for (int c = 0; c < D_ / 32; ++c) {
    v16bf afrag = load_a16(Xbase + c * 32, D_, lane);
#pragma unroll
    for (int t = 0; t < 4; ++t) {
      // B(k,n) = W[d = c*32+k][dk = t*16+n] = WTh[(t*16+n)*D + c*32+k]
      v16bf bfrag = load_bT(WTh + (size_t)(t * 16) * D_ + c * 32, D_, lane);
      acc[t] = wmma_bf16(afrag, bfrag, acc[t]);
    }
  }

  const int rowoff = (lane >> 4) << 3;     // C layout: lanes 16..31 -> rows 8..15
  const int coln   = lane & 15;
  if (proj == 2) {
    // transposed store: Vt[b,h][dk][s]
    __bf16* Obase = Vt + ((size_t)b * H_ + h) * DK_ * S_;
#pragma unroll
    for (int t = 0; t < 4; ++t) {
      const float bia = bias[h * DK_ + t * 16 + coln];
#pragma unroll
      for (int r = 0; r < 8; ++r)
        Obase[(size_t)(t * 16 + coln) * S_ + stile * 16 + rowoff + r] =
            (__bf16)(acc[t][r] + bia);
    }
  } else {
    __bf16* Obase = ((proj == 0) ? Qp : Kp) +
                    (((size_t)b * H_ + h) * S_ + (size_t)stile * 16) * DK_;
#pragma unroll
    for (int t = 0; t < 4; ++t) {
      const float bia = bias[h * DK_ + t * 16 + coln];
#pragma unroll
      for (int r = 0; r < 8; ++r)
        Obase[(size_t)(rowoff + r) * DK_ + t * 16 + coln] =
            (__bf16)(acc[t][r] + bia);
    }
  }
}

// =====================================================================
// Kernel 2: flash-style masked attention.
// Block = 8 waves; wave owns 16 query rows; stream KV in 32-key tiles.
// K is row-major [S][DK] (K^T fragments contiguous); V is [DK][S]
// (P.V fragments contiguous).
// =====================================================================
__global__ __launch_bounds__(256) void attn_kernel(
    const float* __restrict__ masks,
    const __bf16* __restrict__ Qp, const __bf16* __restrict__ Kp,
    const __bf16* __restrict__ Vt, __bf16* __restrict__ Heads) {
  __shared__ alignas(32) __bf16 pshm[8][16 * 40];  // per-wave 16x32 P, stride 40

  const int lane = threadIdx.x & 31;
  const int wave = threadIdx.x >> 5;
  const int bh   = blockIdx.x >> 3;        // blocks: (B*H) * (S/128)
  const int qt   = blockIdx.x & 7;
  const int b    = bh >> 3;
  const int h    = bh & 7;
  const int qbase = qt * 128 + wave * 16;

  const __bf16* Qb = Qp + (((size_t)b * H_ + h) * S_ + qbase) * DK_;
  const __bf16* Kb = Kp + (((size_t)b * H_ + h) * S_) * DK_;     // [S][DK]
  const __bf16* Vb = Vt + (((size_t)b * H_ + h) * DK_) * S_;     // [DK][S]
  const float*  Mb = masks + ((size_t)b * S_ + qbase) * S_;

  // Q fragments for the two 32-wide K-dim chunks of DK=64 (held in VGPRs)
  const v16bf qa0 = load_a16(Qb + 0,  DK_, lane);
  const v16bf qa1 = load_a16(Qb + 32, DK_, lane);

  v8f   o[4] = {};
  float mrow[8], lrow[8];
#pragma unroll
  for (int r = 0; r < 8; ++r) { mrow[r] = -3.0e38f; lrow[r] = 0.0f; }

  __bf16* pb       = pshm[wave];
  const int rowoff = (lane >> 4) << 3;
  const int coln   = lane & 15;

  for (int kv = 0; kv < S_; kv += 32) {
    if (kv + 32 < S_) {                    // hint next K tile toward the caches
      __builtin_prefetch(Kb + (size_t)(kv + 32) * DK_, 0, 1);
      __builtin_prefetch(Vb + (size_t)(kv + 32), 0, 1);
    }

    // ---- scores: S(16x32) = Q(16x64) . K^T(64x32), two 16-col fragments
    v8f s0 = {}, s1 = {};
    {
      v16bf kb;
      kb = load_bT(Kb + (size_t)kv * DK_ + 0,         DK_, lane);
      s0 = wmma_bf16(qa0, kb, s0);
      kb = load_bT(Kb + (size_t)kv * DK_ + 32,        DK_, lane);
      s0 = wmma_bf16(qa1, kb, s0);
      kb = load_bT(Kb + (size_t)(kv + 16) * DK_ + 0,  DK_, lane);
      s1 = wmma_bf16(qa0, kb, s1);
      kb = load_bT(Kb + (size_t)(kv + 16) * DK_ + 32, DK_, lane);
      s1 = wmma_bf16(qa1, kb, s1);
    }

    // ---- scale, multiplicative mask, zero -> -1e9 (faithful to reference)
#pragma unroll
    for (int r = 0; r < 8; ++r) {
      const float m0 = Mb[(size_t)(rowoff + r) * S_ + kv + coln];
      const float m1 = Mb[(size_t)(rowoff + r) * S_ + kv + 16 + coln];
      const float v0 = s0[r] * 0.125f * m0;   // 1/sqrt(64)
      const float v1 = s1[r] * 0.125f * m1;
      s0[r] = (v0 == 0.0f) ? -1.0e9f : v0;
      s1[r] = (v1 == 0.0f) ? -1.0e9f : v1;
    }

    // ---- online softmax (row stats live in 16-lane halves of the wave)
#pragma unroll
    for (int r = 0; r < 8; ++r) {
      float tm = fmaxf(s0[r], s1[r]);
      tm = fmaxf(tm, __shfl_xor(tm, 8, 32));
      tm = fmaxf(tm, __shfl_xor(tm, 4, 32));
      tm = fmaxf(tm, __shfl_xor(tm, 2, 32));
      tm = fmaxf(tm, __shfl_xor(tm, 1, 32));
      const float mnew = fmaxf(mrow[r], tm);
      const float corr = __expf(mrow[r] - mnew);
      const float p0   = __expf(s0[r] - mnew);
      const float p1   = __expf(s1[r] - mnew);
      float ps = p0 + p1;
      ps += __shfl_xor(ps, 8, 32);
      ps += __shfl_xor(ps, 4, 32);
      ps += __shfl_xor(ps, 2, 32);
      ps += __shfl_xor(ps, 1, 32);
      lrow[r] = lrow[r] * corr + ps;
      mrow[r] = mnew;
#pragma unroll
      for (int t = 0; t < 4; ++t) o[t][r] *= corr;
      // stage P in LDS (C layout -> memory) for re-read in A layout
      pb[(rowoff + r) * 40 + coln]      = (__bf16)p0;
      pb[(rowoff + r) * 40 + 16 + coln] = (__bf16)p1;
    }

    // wave-local LDS RAW: LDS ops are in-order per wave, just wait them out
    asm volatile("s_wait_dscnt 0" ::: "memory");
    const v16bf pa = load_a16(pb, 40, lane);
    asm volatile("s_wait_dscnt 0" ::: "memory");  // WAR guard vs next iter's stores

    // ---- O(16x64) += P(16x32) . V(32x64);  B(k,n) = Vt[dk=t*16+n][kv+k]
#pragma unroll
    for (int t = 0; t < 4; ++t) {
      v16bf vb = load_bT(Vb + (size_t)(t * 16) * S_ + kv, S_, lane);
      o[t] = wmma_bf16(pa, vb, o[t]);
    }
  }

  // ---- normalize and store heads in concat layout [B, S, H*DK]
  __bf16* Hb = Heads + ((size_t)b * S_ + qbase) * D_ + (size_t)h * DK_;
#pragma unroll
  for (int t = 0; t < 4; ++t) {
#pragma unroll
    for (int r = 0; r < 8; ++r) {
      const float val = o[t][r] / lrow[r];
      Hb[(size_t)(rowoff + r) * D_ + t * 16 + coln] = (__bf16)val;
    }
  }
}

// =====================================================================
// Kernel 3: output projection  out = heads @ Wo + bo  (f32 out)
// WoT layout: [Dout][Din] -> B fragments contiguous.
// One wave -> 16 rows x 64 output columns.
// =====================================================================
__global__ __launch_bounds__(256) void out_proj_kernel(
    const __bf16* __restrict__ Heads, const __bf16* __restrict__ WoT,
    const float* __restrict__ bo, float* __restrict__ Out) {
  const int lane = threadIdx.x & 31;
  const int wave = threadIdx.x >> 5;
  const int gid  = blockIdx.x * 8 + wave;

  const int ng    = gid & 7;               // 64-col output group
  const int stile = (gid >> 3) & 63;
  const int b     = gid >> 9;

  const __bf16* Hb = Heads + ((size_t)b * S_ + (size_t)stile * 16) * D_;

  v8f acc[4] = {};
  for (int c = 0; c < D_ / 32; ++c) {
    v16bf afrag = load_a16(Hb + c * 32, D_, lane);
#pragma unroll
    for (int t = 0; t < 4; ++t) {
      // B(k,n) = Wo[d = c*32+k][out = ng*64+t*16+n] = WoT[out][d]
      v16bf bfrag =
          load_bT(WoT + (size_t)(ng * 64 + t * 16) * D_ + c * 32, D_, lane);
      acc[t] = wmma_bf16(afrag, bfrag, acc[t]);
    }
  }

  float* Ob        = Out + ((size_t)b * S_ + (size_t)stile * 16) * D_ + ng * 64;
  const int rowoff = (lane >> 4) << 3;
  const int coln   = lane & 15;
#pragma unroll
  for (int t = 0; t < 4; ++t) {
    const float bia = bo[ng * 64 + t * 16 + coln];
#pragma unroll
    for (int r = 0; r < 8; ++r)
      Ob[(size_t)(rowoff + r) * D_ + t * 16 + coln] = acc[t][r] + bia;
  }
}

// =====================================================================
extern "C" void kernel_launch(void* const* d_in, const int* in_sizes, int n_in,
                              void* d_out, int out_size, void* d_ws, size_t ws_size,
                              hipStream_t stream) {
  const float* query = (const float*)d_in[0];
  const float* value = (const float*)d_in[1];
  const float* masks = (const float*)d_in[2];
  const float* Wq    = (const float*)d_in[3];
  const float* bq    = (const float*)d_in[4];
  const float* Wk    = (const float*)d_in[5];
  const float* bk    = (const float*)d_in[6];
  const float* Wv    = (const float*)d_in[7];
  const float* bv    = (const float*)d_in[8];
  const float* Wo    = (const float*)d_in[9];
  const float* bo    = (const float*)d_in[10];
  float* out = (float*)d_out;

  const size_t actElems  = (size_t)B_ * S_ * D_;         // 8M
  const size_t projElems = (size_t)B_ * H_ * S_ * DK_;   // 8M
  const size_t wElems    = (size_t)H_ * D_ * DK_;        // 256K
  const size_t woElems   = (size_t)D_ * D_;              // 256K

  __bf16* Xq    = (__bf16*)d_ws;
  __bf16* Xv    = Xq + actElems;
  __bf16* WqT   = Xv + actElems;
  __bf16* WkT   = WqT + wElems;
  __bf16* WvT   = WkT + wElems;
  __bf16* WoT   = WvT + wElems;
  __bf16* Qp    = WoT + woElems;
  __bf16* Kp    = Qp + projElems;
  __bf16* Vt    = Kp + projElems;            // transposed [B,H][DK][S]
  __bf16* Heads = Vt + projElems;            // [B,S,D]

  // ---- one-time precision/layout conversion (tiny vs. GEMM cost)
  cvt_kernel<<<actElems / 8 / 256, 256, 0, stream>>>(query, Xq);
  cvt_kernel<<<actElems / 8 / 256, 256, 0, stream>>>(value, Xv);
  wtrans_kernel<<<wElems / 256, 256, 0, stream>>>(Wq, WqT, D_, DK_);
  wtrans_kernel<<<wElems / 256, 256, 0, stream>>>(Wk, WkT, D_, DK_);
  wtrans_kernel<<<wElems / 256, 256, 0, stream>>>(Wv, WvT, D_, DK_);
  wtrans_kernel<<<woElems / 256, 256, 0, stream>>>(Wo, WoT, D_, D_);

  // ---- 3 projections * B*H*(S/16) wave-tiles / 8 waves per block
  qkv_proj_kernel<<<3 * B_ * H_ * (S_ / 16) / 8, 256, 0, stream>>>(
      Xq, Xv, WqT, WkT, WvT, bq, bk, bv, Qp, Kp, Vt);

  attn_kernel<<<B_ * H_ * (S_ / 128), 256, 0, stream>>>(masks, Qp, Kp, Vt, Heads);

  out_proj_kernel<<<B_ * (S_ / 16) * (D_ / 64) / 8, 256, 0, stream>>>(
      Heads, WoT, bo, out);
}
